// Attention_61168924229643
// MI455X (gfx1250) — matrix operand order repaired and checked by reading, not compiled, not run on hardware
//
#include <hip/hip_runtime.h>

// ---------------- types ----------------
typedef _Float16 v16h __attribute__((ext_vector_type(16)));
typedef _Float16 v8h  __attribute__((ext_vector_type(8)));
typedef float    v8f  __attribute__((ext_vector_type(8)));
typedef unsigned int u32x4 __attribute__((ext_vector_type(4)));
typedef int          i32x4 __attribute__((ext_vector_type(4)));
typedef int          i32x8 __attribute__((ext_vector_type(8)));

#define HEADS 4
#define DH    32
#define CDIM  128
#define NTOK  1024
#define BATCH 8
#define ROWS  (BATCH * NTOK)   // 8192

__device__ __forceinline__ v8f zero_v8f() {
    v8f z;
    for (int i = 0; i < 8; ++i) z[i] = 0.0f;
    return z;
}

// A-fragment (16x32, f16). Per CDNA5 ISA 7.12.2:
// lanes 0-15 hold rows 0-15 with K = {0..7, 16..23}; lanes 16-31 same rows, K = {8..15, 24..31}.
__device__ __forceinline__ v16h load_a_frag(const _Float16* base, int ld) {
    int lane = threadIdx.x & 31;
    int r = lane & 15, g = lane >> 4;
    const _Float16* p = base + r * ld + g * 8;
    v8h lo = *(const v8h*)p;          // K = g*8 .. g*8+7
    v8h hi = *(const v8h*)(p + 16);   // K = 16+g*8 .. 16+g*8+7
    return __builtin_shufflevector(lo, hi, 0,1,2,3,4,5,6,7,8,9,10,11,12,13,14,15);
}

// B-fragment (32x16, f16) from a B^T layout: BT[n][k], row stride ld.
// lanes 0-15 hold column n with K = {0..7, 16..23}; lanes 16-31 same column, K = {8..15, 24..31}.
__device__ __forceinline__ v16h load_b_frag(const _Float16* base, int ld) {
    int lane = threadIdx.x & 31;
    int n = lane & 15, g = lane >> 4;
    const _Float16* p = base + n * ld + g * 8;
    v8h lo = *(const v8h*)p;          // K = g*8 .. g*8+7
    v8h hi = *(const v8h*)(p + 16);   // K = 16+g*8 .. 16+g*8+7
    return __builtin_shufflevector(lo, hi, 0,1,2,3,4,5,6,7,8,9,10,11,12,13,14,15);
}

__device__ __forceinline__ float grp16_max(float v) {
    v = fmaxf(v, __shfl_xor(v, 1, 32));
    v = fmaxf(v, __shfl_xor(v, 2, 32));
    v = fmaxf(v, __shfl_xor(v, 4, 32));
    v = fmaxf(v, __shfl_xor(v, 8, 32));
    return v;
}
__device__ __forceinline__ float grp16_sum(float v) {
    v += __shfl_xor(v, 1, 32);
    v += __shfl_xor(v, 2, 32);
    v += __shfl_xor(v, 4, 32);
    v += __shfl_xor(v, 8, 32);
    return v;
}

// ---------------- Tensor Data Mover: 2-D tile load, f16 elements ----------------
// Builds D# group0/group1 per CDNA5 ISA 8.3/8.4 and issues TENSOR_LOAD_TO_LDS.
// tile_d0 = contiguous elements per row, tile_d1 = rows, stride0 = elements
// between rows in global memory. LDS destination is packed row-major.
__device__ __forceinline__ void tdm_load_2d_f16(uint32_t lds_off, const void* gsrc,
                                                uint32_t tile_d0, uint32_t tile_d1,
                                                uint32_t stride0)
{
    uint64_t ga = (uint64_t)(uintptr_t)gsrc;
    u32x4 g0;
    g0[0] = 1u;                                               // count=1, user mode
    g0[1] = lds_off;                                          // lds_addr (bytes)
    g0[2] = (uint32_t)ga;                                     // global_addr[31:0]
    g0[3] = (uint32_t)((ga >> 32) & 0x01FFFFFFu) | (2u << 30);// addr[56:32] | type=2
    // group1: tensor dims == tile dims (no OOB fill), data_size=1 (2 bytes)
    i32x8 g1;
    g1[0] = 0x00010000;                                       // data_size=2B, mask=0
    g1[1] = (int)((tile_d0 & 0xFFFFu) << 16);                 // tensor_dim0[15:0]
    g1[2] = (int)(((tile_d0 >> 16) & 0xFFFFu) |
                  ((tile_d1 & 0xFFFFu) << 16));               // td0 hi | td1 lo
    g1[3] = (int)(((tile_d1 >> 16) & 0xFFFFu) |
                  ((tile_d0 & 0xFFFFu) << 16));               // td1 hi | tile_dim0
    g1[4] = (int)(tile_d1 & 0xFFFFu);                         // tile_dim1, tile_dim2=0
    g1[5] = (int)stride0;                                     // tensor_dim0_stride lo
    g1[6] = 0;                                                // stride0 hi | stride1 lo
    g1[7] = 0;                                                // stride1 hi
    i32x4 z4 = {0, 0, 0, 0};
#if __has_include(<hip/amd_detail/amd_gfx1250_TDM.h>)
    i32x8 z8 = {0, 0, 0, 0, 0, 0, 0, 0};
    __builtin_amdgcn_tensor_load_to_lds(g0, g1, z4, z4, z8, 0);
#else
    __builtin_amdgcn_tensor_load_to_lds(g0, g1, z4, z4, 0);
#endif
}

__device__ __forceinline__ uint32_t lds_offset(const void* p) {
    return (uint32_t)(uintptr_t)p;   // low 32 bits of generic addr = LDS offset
}

// ---------------- K1: QKV projection (x @ w_qkv -> f16 q/k, transposed v) ------
// grid (128, 3): 64-row tiles x {q,k,v} col-block of 128. 128 threads = 4 waves.
__global__ __launch_bounds__(128) void qkv_kernel(
    const float* __restrict__ x, const float* __restrict__ wqkv,
    _Float16* __restrict__ qh, _Float16* __restrict__ kh, _Float16* __restrict__ vtg)
{
    __shared__ alignas(16) _Float16 Xs[64 * 128];    // row-major f16 x tile
    __shared__ alignas(16) _Float16 WT[128 * 128];   // WT[n][k] = w[k][cb*128+n]
    const int rt = blockIdx.x, cb = blockIdx.y;
    const int tid = threadIdx.x, wave = tid >> 5, lane = tid & 31;
    const int ln = lane & 15, lg = lane >> 4;

    for (int e = tid; e < 64 * 128; e += 128) {
        int r = e >> 7, k = e & 127;
        Xs[e] = (_Float16)x[(size_t)(rt * 64 + r) * CDIM + k];
    }
    for (int e = tid; e < 128 * 128; e += 128) {
        int n = e >> 7, k = e & 127;
        WT[n * 128 + k] = (_Float16)wqkv[(size_t)k * (3 * CDIM) + cb * CDIM + n];
    }
    __syncthreads();

    v8f acc[8];
    for (int t = 0; t < 8; ++t) acc[t] = zero_v8f();
    for (int ks = 0; ks < 4; ++ks) {
        v16h aX = load_a_frag(&Xs[wave * 16 * 128] + ks * 32, 128);
        for (int t = 0; t < 8; ++t) {
            v16h bW = load_b_frag(&WT[t * 16 * 128] + ks * 32, 128);
            acc[t] = __builtin_amdgcn_wmma_f32_16x16x32_f16(false, aX, false, bW,
                                                            (short)0, acc[t], false, false);
        }
    }
    if (cb < 2) {
        _Float16* out = (cb == 0) ? qh : kh;
        for (int t = 0; t < 8; ++t)
            for (int r = 0; r < 8; ++r) {
                int row = rt * 64 + wave * 16 + lg * 8 + r;
                out[(size_t)row * CDIM + t * 16 + ln] = (_Float16)acc[t][r];
            }
    } else {
        // V: store transposed per (b,h): vtg[((b*HEADS+h)*DH + d)*NTOK + tok]
        for (int t = 0; t < 8; ++t)
            for (int r = 0; r < 8; ++r) {
                int row = rt * 64 + wave * 16 + lg * 8 + r;
                int bb = row >> 10, tok = row & (NTOK - 1);
                int col = t * 16 + ln, hh = col >> 5, d = col & 31;
                vtg[(size_t)((bb * HEADS + hh) * DH + d) * NTOK + tok] = (_Float16)acc[t][r];
            }
    }
}

// ---------------- K2: token-axis L2 norms (faithful axis=1 quirk) ----------------
__global__ void norm_zero(float* nq, float* nk) {
    int t = blockIdx.x * 256 + threadIdx.x;
    if (t < BATCH * CDIM) { nq[t] = 0.0f; nk[t] = 0.0f; }
}

__global__ __launch_bounds__(256) void norm_partial(
    const _Float16* __restrict__ qh, const _Float16* __restrict__ kh,
    float* __restrict__ nq, float* __restrict__ nk)
{
    int b  = blockIdx.x >> 3;
    int rc = blockIdx.x & 7;
    int t  = threadIdx.x;
    const _Float16* src = (t < 128) ? qh : kh;
    float* dst          = (t < 128) ? nq : nk;
    int c = t & 127;
    float s = 0.0f;
    for (int j = 0; j < 128; ++j) {
        float v = (float)src[(size_t)(b * NTOK + rc * 128 + j) * CDIM + c];
        s += v * v;
    }
    atomicAdd(&dst[b * CDIM + c], s);
}

__global__ void make_scales(const float* __restrict__ nq, const float* __restrict__ nk,
                            float* __restrict__ sq, float* __restrict__ sk)
{
    int t = blockIdx.x * 256 + threadIdx.x;
    if (t < BATCH * CDIM) {
        sq[t] = 10.0f / fmaxf(sqrtf(nq[t]), 1e-12f);  // SCALE folded into q
        sk[t] = 1.0f  / fmaxf(sqrtf(nk[t]), 1e-12f);
    }
}

// Pre-scale q/k in place so attention staging is a pure TDM copy.
__global__ __launch_bounds__(256) void scale_qk(
    _Float16* __restrict__ qh, _Float16* __restrict__ kh,
    const float* __restrict__ sq, const float* __restrict__ sk)
{
    size_t i = (size_t)blockIdx.x * 256 + threadIdx.x;  // over ROWS*CDIM
    int c = (int)(i & (CDIM - 1));
    int b = (int)(i >> 17);                              // / (NTOK*CDIM)
    qh[i] = (_Float16)((float)qh[i] * sq[b * CDIM + c]);
    kh[i] = (_Float16)((float)kh[i] * sk[b * CDIM + c]);
}

// ---------------- K3: fused flash attention per (b, h, 128-row tile) ----------------
// softmax(s - argmax_idx) == softmax(s): per-row constant shift is invariant.
// All operand staging goes through the Tensor Data Mover (TENSORcnt).
__global__ __launch_bounds__(256) void attn_kernel(
    const _Float16* __restrict__ qs16, const _Float16* __restrict__ ks16,
    const _Float16* __restrict__ vtg, _Float16* __restrict__ oh)
{
    __shared__ alignas(16) _Float16 Qs[128 * 32];      // scaled Q tile, row-major
    __shared__ alignas(16) _Float16 Ks[128 * 32];      // scaled K block (== B^T for S)
    __shared__ alignas(16) _Float16 VT[32 * 128];      // V block transposed: VT[d][j]
    __shared__ alignas(16) _Float16 Ps[8][16 * 128];   // per-wave P strip, row-major

    const int b = blockIdx.x, h = blockIdx.y, rt = blockIdx.z;
    const int tid = threadIdx.x, wave = tid >> 5, lane = tid & 31;
    const int ln = lane & 15, lg = lane >> 4;

    const _Float16* qbase = qs16 + (size_t)(b * NTOK + rt * 128) * CDIM + h * DH;
    const _Float16* kbase = ks16 + (size_t)(b * NTOK) * CDIM + h * DH;
    const _Float16* vbase = vtg + (size_t)((b * HEADS + h) * DH) * NTOK;

    // TDM: Q tile (128 rows x 32 cols, row stride CDIM) -> packed LDS
    if (wave == 0) {
        tdm_load_2d_f16(lds_offset(Qs), qbase, DH, 128, CDIM);
        __builtin_amdgcn_s_wait_tensorcnt((short)0);
    }
    __syncthreads();

    v16h aQ = load_a_frag(&Qs[wave * 16 * 32], 32);

    v8f O0 = zero_v8f(), O1 = zero_v8f();
    float mrow[8], lrow[8];
    for (int r = 0; r < 8; ++r) { mrow[r] = -__builtin_inff(); lrow[r] = 0.0f; }

    for (int jb = 0; jb < 8; ++jb) {
        __syncthreads();   // previous iteration finished reading Ks/VT
        if (wave == 0) {
            // K block: 128 rows x 32 cols; V^T block: 32 rows x 128 cols (stride NTOK)
            tdm_load_2d_f16(lds_offset(Ks), kbase + (size_t)(jb * 128) * CDIM, DH, 128, CDIM);
            tdm_load_2d_f16(lds_offset(VT), vbase + jb * 128, 128, DH, NTOK);
            __builtin_amdgcn_s_wait_tensorcnt((short)0);
        }
        if (jb + 1 < 8)
            __builtin_prefetch(kbase + (size_t)((jb + 1) * 128) * CDIM, 0, 1);
        __syncthreads();

        // S strip: 16 x 128, K=32 in one WMMA per 16-col tile
        v8f S[8];
        for (int t = 0; t < 8; ++t) {
            v16h bK = load_b_frag(&Ks[t * 16 * 32], 32);
            S[t] = __builtin_amdgcn_wmma_f32_16x16x32_f16(false, aQ, false, bK,
                                                          (short)0, zero_v8f(), false, false);
        }

        // online softmax; lane owns rows m = lg*8 + r (matches C/D layout)
        float mnew[8], corr[8], rsum[8];
        for (int r = 0; r < 8; ++r) {
            float mx = S[0][r];
            for (int t = 1; t < 8; ++t) mx = fmaxf(mx, S[t][r]);
            mx = grp16_max(mx);
            float mn = fmaxf(mrow[r], mx);
            mnew[r] = mn;
            corr[r] = __expf(mrow[r] - mn);
            rsum[r] = 0.0f;
        }
        _Float16* Pw = &Ps[wave][0];
        for (int t = 0; t < 8; ++t)
            for (int r = 0; r < 8; ++r) {
                float p = __expf(S[t][r] - mnew[r]);
                rsum[r] += p;
                Pw[(lg * 8 + r) * 128 + t * 16 + ln] = (_Float16)p;
            }
        for (int r = 0; r < 8; ++r) {
            float rs = grp16_sum(rsum[r]);
            lrow[r] = lrow[r] * corr[r] + rs;
            mrow[r] = mnew[r];
            O0[r] *= corr[r];
            O1[r] *= corr[r];
        }

        // O += P (16x128) x V (128x32), split K into 4 chunks of 32
        for (int ks = 0; ks < 4; ++ks) {
            v16h aP  = load_a_frag(Pw + ks * 32, 128);
            v16h bv0 = load_b_frag(&VT[0]        + ks * 32, 128);  // out cols 0..15
            v16h bv1 = load_b_frag(&VT[16 * 128] + ks * 32, 128);  // out cols 16..31
            O0 = __builtin_amdgcn_wmma_f32_16x16x32_f16(false, aP, false, bv0,
                                                        (short)0, O0, false, false);
            O1 = __builtin_amdgcn_wmma_f32_16x16x32_f16(false, aP, false, bv1,
                                                        (short)0, O1, false, false);
        }
    }

    // normalize and write: layout (b, token, h*32+d) == reference transpose+reshape
    for (int r = 0; r < 8; ++r) {
        float inv = 1.0f / lrow[r];
        int row = rt * 128 + wave * 16 + lg * 8 + r;
        size_t base = (size_t)(b * NTOK + row) * CDIM + h * DH;
        oh[base + ln]      = (_Float16)(O0[r] * inv);
        oh[base + 16 + ln] = (_Float16)(O1[r] * inv);
    }
}

// ---------------- K4: output projection (attn_out @ w_out + b_out -> f32) ----------------
__global__ __launch_bounds__(128) void proj_kernel(
    const _Float16* __restrict__ oh, const float* __restrict__ wout,
    const float* __restrict__ bout, float* __restrict__ out)
{
    __shared__ alignas(16) _Float16 As[64 * 128];
    __shared__ alignas(16) _Float16 WT[128 * 128];   // WT[n][k] = w_out[k][n]
    const int rt = blockIdx.x;
    const int tid = threadIdx.x, wave = tid >> 5, lane = tid & 31;
    const int ln = lane & 15, lg = lane >> 4;

    for (int e = tid; e < 64 * 128; e += 128)
        As[e] = oh[(size_t)(rt * 64) * CDIM + e];
    for (int e = tid; e < 128 * 128; e += 128) {
        int n = e >> 7, k = e & 127;
        WT[n * 128 + k] = (_Float16)wout[(size_t)k * CDIM + n];
    }
    __syncthreads();

    v8f acc[8];
    for (int t = 0; t < 8; ++t) acc[t] = zero_v8f();
    for (int ks = 0; ks < 4; ++ks) {
        v16h aA = load_a_frag(&As[wave * 16 * 128] + ks * 32, 128);
        for (int t = 0; t < 8; ++t) {
            v16h bW = load_b_frag(&WT[t * 16 * 128] + ks * 32, 128);
            acc[t] = __builtin_amdgcn_wmma_f32_16x16x32_f16(false, aA, false, bW,
                                                            (short)0, acc[t], false, false);
        }
    }
    for (int t = 0; t < 8; ++t)
        for (int r = 0; r < 8; ++r) {
            int row = rt * 64 + wave * 16 + lg * 8 + r;
            int col = t * 16 + ln;
            out[(size_t)row * CDIM + col] = acc[t][r] + bout[col];
        }
}

// ---------------- launcher ----------------
extern "C" void kernel_launch(void* const* d_in, const int* in_sizes, int n_in,
                              void* d_out, int out_size, void* d_ws, size_t ws_size,
                              hipStream_t stream) {
    const float* x    = (const float*)d_in[0];
    const float* wqkv = (const float*)d_in[1];
    const float* wout = (const float*)d_in[2];
    const float* bout = (const float*)d_in[3];
    float* out = (float*)d_out;

    // workspace layout
    _Float16* qh  = (_Float16*)d_ws;                 // 8192*128 f16 = 2 MB
    _Float16* kh  = qh  + (size_t)ROWS * CDIM;
    _Float16* vtg = kh  + (size_t)ROWS * CDIM;       // (B,H,DH,NTOK) transposed V
    _Float16* oh  = vtg + (size_t)ROWS * CDIM;
    float* nq = (float*)(oh + (size_t)ROWS * CDIM);  // 1024 f32 each
    float* nk = nq + BATCH * CDIM;
    float* sq = nk + BATCH * CDIM;
    float* sk = sq + BATCH * CDIM;

    qkv_kernel  <<<dim3(128, 3), 128, 0, stream>>>(x, wqkv, qh, kh, vtg);
    norm_zero   <<<4, 256, 0, stream>>>(nq, nk);
    norm_partial<<<64, 256, 0, stream>>>(qh, kh, nq, nk);
    make_scales <<<4, 256, 0, stream>>>(nq, nk, sq, sk);
    scale_qk    <<<(ROWS * CDIM) / 256, 256, 0, stream>>>(qh, kh, sq, sk);
    attn_kernel <<<dim3(BATCH, HEADS, 8), 256, 0, stream>>>(qh, kh, vtg, oh);
    proj_kernel <<<128, 128, 0, stream>>>(oh, wout, bout, out);
}
